// C3_layer_51616916963982
// MI455X (gfx1250) — matrix operand order repaired and checked by reading, not compiled, not run on hardware
//
#include <hip/hip_runtime.h>

typedef __attribute__((ext_vector_type(16))) _Float16 v16h;
typedef __attribute__((ext_vector_type(8)))  float    v8f;
typedef __attribute__((ext_vector_type(4)))  float    v4f;

#define OC     16
#define CIN    6
#define HW     256
#define OWH    252
#define K5     5
#define NCHUNK 8     // K = 256 = 8 chunks of 32 (30 real tap-rows * 8-padded kx)
#define TRMAX  30    // real tap rows (c*5+ky)

// block tile: 16 output rows x 32 output cols x 16 channels
#define BY  16
#define BX  32
#define LR  20       // staged input rows  (BY + 4 halo)
#define LC  40       // staged input cols  (max read slot is 38)
#define LCP 48       // padded LDS row stride (halves) -> bank-friendly

__device__ __forceinline__ int iminc(int a, int b) { return a < b ? a : b; }

static __device__ const int dCH3[6][3] = {{0,1,2},{1,2,3},{2,3,4},{3,4,5},{0,4,5},{0,1,5}};
static __device__ const int dCH4[9][4] = {{0,1,2,3},{1,2,3,4},{2,3,4,5},{0,3,4,5},{0,1,4,5},
                                          {0,1,2,5},{0,1,3,4},{1,2,4,5},{0,2,3,5}};

// Pack dense (16 x 256) f16 weight matrix in WMMA B-fragment order:
// for chunk ch, lane l: halves j=0..15 hold column n=l&15, K = ch*32 + (l<16?0:16) + j.
__global__ void prep_weights(const float* __restrict__ w3, const float* __restrict__ b3,
                             const float* __restrict__ w4, const float* __restrict__ b4,
                             const float* __restrict__ w6, const float* __restrict__ b6,
                             _Float16* __restrict__ Wpk, float* __restrict__ biasp)
{
    int t    = threadIdx.x;            // 0..255
    int ch   = t >> 5;
    int lane = t & 31;
    int n    = lane & 15;
    int kb   = ch * 32 + ((lane & 16) ? 16 : 0);
    _Float16* dst = Wpk + (ch * 32 + lane) * 16;
    for (int j = 0; j < 16; ++j) {
        int K  = kb + j;
        int tr = K >> 3;
        int kx = K & 7;
        float v = 0.0f;
        if (tr < TRMAX && kx < K5) {
            int c = tr / K5, ky = tr % K5;
            if (n < 6) {
                for (int i = 0; i < 3; ++i)
                    if (dCH3[n][i] == c) v = w3[((n * 3 + i) * K5 + ky) * K5 + kx];
            } else if (n < 15) {
                int g = n - 6;
                for (int i = 0; i < 4; ++i)
                    if (dCH4[g][i] == c) v = w4[((g * 4 + i) * K5 + ky) * K5 + kx];
            } else {
                v = w6[(c * K5 + ky) * K5 + kx];
            }
        }
        dst[j] = (_Float16)v;
    }
    if (t < OC) {
        float bv = (t < 6) ? b3[t] : (t < 15) ? b4[t - 6] : b6[0];
        biasp[t] = bv;
    }
}

__global__ __launch_bounds__(256)
void gconv_wmma(const float* __restrict__ x, const _Float16* __restrict__ Wpk,
                const float* __restrict__ biasp, float* __restrict__ out, int B)
{
    // two phase-shifted f16 copies: lds[1][..][s] = x(col s+1), so any even/odd
    // column pair is a dword-aligned LDS read in one of the two copies.
    __shared__ _Float16 lds[2][CIN][LR][LCP];

    const int tid  = threadIdx.x;
    const int lane = tid & 31;
    const int wv   = tid >> 5;
    const int x0   = blockIdx.x * BX;
    const int y0   = blockIdx.y * BY;
    const int b    = blockIdx.z;

    // ---- stage input tile, f32 -> f16 (regular-temporal: halo rows get L2 reuse) ----
    const float* xb = x + (size_t)b * CIN * HW * HW;
    for (int idx = tid; idx < CIN * LR * LC; idx += 256) {
        int col = idx % LC;
        int r   = (idx / LC) % LR;
        int c   = idx / (LC * LR);
        int gy  = iminc(y0 + r, HW - 1);     // clamped rows/cols only feed
        int gx  = iminc(x0 + col, HW - 1);   // zero weights or masked outputs
        _Float16 h = (_Float16)xb[(c * HW + gy) * HW + gx];
        lds[0][c][r][col] = h;
        if (col > 0) lds[1][c][r][col - 1] = h;
    }
    __syncthreads();

    // ---- load B fragments (held in registers for the whole block) ----
    v16h bfr[NCHUNK];
#pragma unroll
    for (int ch = 0; ch < NCHUNK; ++ch)
        bfr[ch] = *(const v16h*)(Wpk + (ch * 32 + lane) * 16);

    const float bias = biasp[lane & 15];
    const int   m    = lane & 15;
    const bool  hi   = (lane & 16) != 0;

#pragma unroll
    for (int t = 0; t < 4; ++t) {
        const int dy   = wv * 2 + (t & 1);   // 0..15
        const int xoff = (t >> 1) * 16;      // 0 or 16
        const int y    = y0 + dy;

        v8f acc;
#pragma unroll
        for (int v = 0; v < 8; ++v) acc[v] = bias;

        const int bc  = xoff + m;            // lane's base input column
        const int par = bc & 1;
        const _Float16* lbase = &lds[par][0][0][0] + (bc - par); // dword-aligned

#pragma unroll
        for (int ch = 0; ch < NCHUNK; ++ch) {
            int trA = ch * 4 + (hi ? 2 : 0); // lanes 0-15: K 0-7 | lanes 16-31: K 16-23
            int trB = trA + 1;               // K 8-15 | 24-31
            if (trA >= TRMAX) trA = TRMAX - 1;  // padded tap rows: zero weights
            if (trB >= TRMAX) trB = TRMAX - 1;
            const int cA = trA / K5, kyA = trA % K5;
            const int cB = trB / K5, kyB = trB % K5;
            const unsigned int* pA =
                (const unsigned int*)(lbase + (cA * LR + dy + kyA) * LCP);
            const unsigned int* pB =
                (const unsigned int*)(lbase + (cB * LR + dy + kyB) * LCP);
            union { v16h h; unsigned int u[8]; } af;
            af.u[0] = pA[0]; af.u[1] = pA[1]; af.u[2] = pA[2]; af.u[3] = pA[3];
            af.u[4] = pB[0]; af.u[5] = pB[1]; af.u[6] = pB[2]; af.u[7] = pB[3];
            acc = __builtin_amdgcn_wmma_f32_16x16x32_f16(
                false, af.h, false, bfr[ch], (short)0, acc, false, false);
        }

        // ---- store: lane covers N=lane&15, M = v (+8 for hi lanes) ----
        // Output is write-once streaming (520 MB of the 721 MB bill): store NT
        // so it doesn't thrash input halo lines out of the 192 MB L2.
        if (y < OWH) {
            const int xb0 = x0 + xoff + (hi ? 8 : 0);
            float* orow = out + (((size_t)b * OC + (lane & 15)) * OWH + y) * OWH;
            if (xb0 + 7 < OWH) {
                v4f lo  = {acc[0], acc[1], acc[2], acc[3]};
                v4f hi4 = {acc[4], acc[5], acc[6], acc[7]};
                __builtin_nontemporal_store(lo,  (v4f*)(orow + xb0));
                __builtin_nontemporal_store(hi4, (v4f*)(orow + xb0 + 4));
            } else {
#pragma unroll
                for (int v = 0; v < 8; ++v) {
                    int xx = xb0 + v;
                    if (xx < OWH) __builtin_nontemporal_store(acc[v], orow + xx);
                }
            }
        }
    }
}

extern "C" void kernel_launch(void* const* d_in, const int* in_sizes, int n_in,
                              void* d_out, int out_size, void* d_ws, size_t ws_size,
                              hipStream_t stream)
{
    const float* x  = (const float*)d_in[0];
    const float* w3 = (const float*)d_in[1];
    const float* b3 = (const float*)d_in[2];
    const float* w4 = (const float*)d_in[3];
    const float* b4 = (const float*)d_in[4];
    const float* w6 = (const float*)d_in[5];
    const float* b6 = (const float*)d_in[6];

    const int B = in_sizes[0] / (CIN * HW * HW);   // 128

    _Float16* Wpk   = (_Float16*)d_ws;                                  // 8 KB
    float*    biasp = (float*)((char*)d_ws + NCHUNK * 32 * 16 * 2);     // +64 B

    prep_weights<<<1, 256, 0, stream>>>(w3, b3, w4, b4, w6, b6, Wpk, biasp);

    dim3 grid((OWH + BX - 1) / BX, (OWH + BY - 1) / BY, B);  // 8 x 16 x 128
    gconv_wmma<<<grid, 256, 0, stream>>>(x, Wpk, biasp, (float*)d_out, B);
}
